// MultiHeadAttention_26113401160042
// MI455X (gfx1250) — matrix operand order repaired and checked by reading, not compiled, not run on hardware
//
#include <hip/hip_runtime.h>
#include <hip/hip_bf16.h>

// ---------------- types ----------------
typedef __bf16 bf16_t;
typedef __bf16 v16bf __attribute__((ext_vector_type(16)));
typedef __bf16 v8bf  __attribute__((ext_vector_type(8)));
typedef float  v8f   __attribute__((ext_vector_type(8)));

union ABFrag { v16bf v; v8bf half8[2]; };

#define D_MODEL 1024
#define N_HEADS 16
#define DQ      64
#define BATCH   2
#define SEQ     2048
#define MROWS   (BATCH*SEQ)          // 4096 token rows
#define LN_EPS  1e-5f

static __device__ __forceinline__ v8f wmma_bf16(v16bf a, v16bf b, v8f c) {
    return __builtin_amdgcn_wmma_f32_16x16x32_bf16(false, a, false, b, (short)0, c, false, false);
}

// ---------------- stage 0: fp32 -> bf16 ----------------
__global__ void cvt_f32_bf16(const float* __restrict__ src, bf16_t* __restrict__ dst, int n) {
    int i = blockIdx.x * blockDim.x + threadIdx.x;
    int stride = gridDim.x * blockDim.x;
    for (; i < n; i += stride) dst[i] = (bf16_t)src[i];
}

// ---------------- stage 1: projections Y = X @ W^T (bf16 WMMA, fp32 acc) ----
// mode z: 0 -> Qp[b,h,s,d], 1 -> Kp[b,h,s,d], 2 -> Vt[b,h,d,s] (transposed!)
__global__ void __launch_bounds__(256) proj_wmma(
    const bf16_t* __restrict__ Xq, const bf16_t* __restrict__ Xk, const bf16_t* __restrict__ Xv,
    const bf16_t* __restrict__ Wq, const bf16_t* __restrict__ Wk, const bf16_t* __restrict__ Wv,
    bf16_t* __restrict__ Qp, bf16_t* __restrict__ Kp, bf16_t* __restrict__ Vt)
{
    const int mode = blockIdx.z;
    const bf16_t* X = (mode == 0) ? Xq : (mode == 1) ? Xk : Xv;
    const bf16_t* W = (mode == 0) ? Wq : (mode == 1) ? Wk : Wv;

    const int tid  = threadIdx.x;
    const int lane = tid & 31;
    const int wid  = tid >> 5;
    const int hi   = lane >> 4;     // half-wave select
    const int lo   = lane & 15;

    const int m0 = blockIdx.y * 64 + (wid >> 1) * 16;   // token rows
    const int n0 = blockIdx.x * 32 + (wid & 1) * 16;    // output features

    const bf16_t* arow = X + (size_t)(m0 + lo) * D_MODEL;
    const bf16_t* brow = W + (size_t)(n0 + lo) * D_MODEL;

    v8f acc = {0.f,0.f,0.f,0.f,0.f,0.f,0.f,0.f};
    for (int k0 = 0; k0 < D_MODEL; k0 += 32) {
        ABFrag a;
        a.half8[0] = *(const v8bf*)(arow + k0 + hi * 16);       // K = 0..7 / 16..23
        a.half8[1] = *(const v8bf*)(arow + k0 + 8 + hi * 16);   // K = 8..15 / 24..31
        v16bf bfr  = *(const v16bf*)(brow + k0 + hi * 16);      // col = n0+lo, K span by half
        acc = wmma_bf16(a.v, bfr, acc);
    }

    // C layout: VGPR r -> row m0 + r + 8*hi, col n0 + lo
    for (int r = 0; r < 8; ++r) {
        int m = m0 + r + hi * 8;
        int n = n0 + lo;
        int bb = m >> 11;           // / SEQ
        int s  = m & (SEQ - 1);
        int h  = n >> 6;            // / DQ
        int d  = n & (DQ - 1);
        bf16_t val = (bf16_t)acc[r];
        if (mode == 2) {
            Vt[(((size_t)(bb * N_HEADS + h)) * DQ + d) * SEQ + s] = val;
        } else {
            bf16_t* P = (mode == 0) ? Qp : Kp;
            P[(((size_t)(bb * N_HEADS + h)) * SEQ + s) * DQ + d] = val;
        }
    }
}

// ---------------- stage 2: attention (softmax + attn write + context) -------
// one block = (b, h, 16 query rows); 8 waves each own 256 keys
__global__ void __launch_bounds__(256) attn_wmma(
    const bf16_t* __restrict__ Qp, const bf16_t* __restrict__ Kp, const bf16_t* __restrict__ Vt,
    float* __restrict__ attn_out, float* __restrict__ ctx_out)
{
    __shared__ float wave_m[8][16];
    __shared__ float wave_s[8][16];
    __shared__ float gM[16];
    __shared__ float gRinv[16];
    __shared__ __align__(64) bf16_t stageQ[16 * DQ];    // 2 KB: async-staged Q tile
    __shared__ __align__(32) bf16_t stage[8][16][32];   // per-wave prob tile re-layout
    __shared__ __align__(16) float ctxLds[16][DQ];      // 4 KB: reduced context

    const int tid  = threadIdx.x;
    const int lane = tid & 31;
    const int wid  = tid >> 5;
    const int hi   = lane >> 4;
    const int lo   = lane & 15;

    const int idx = blockIdx.x;
    const int q0  = (idx & 127) << 4;       // query tile start
    const int h   = (idx >> 7) & 15;
    const int b   = idx >> 11;

    const bf16_t* Qh = Qp + (size_t)(b * N_HEADS + h) * SEQ * DQ;
    const bf16_t* Kh = Kp + (size_t)(b * N_HEADS + h) * SEQ * DQ;
    const bf16_t* Vh = Vt + (size_t)(b * N_HEADS + h) * DQ * SEQ;
    float* attnB = attn_out + ((size_t)(b * N_HEADS + h) * SEQ + q0) * SEQ;

    for (int i = tid; i < 16 * DQ; i += 256) ((float*)ctxLds)[i] = 0.f;

    // ---- async-DMA the contiguous 2KB Q tile into LDS (wave 0 only) --------
    // GLOBAL_LOAD_ASYNC_TO_LDS_B128: per-lane 16B global -> LDS, ASYNCcnt-tracked
    if (wid == 0) {
        const char* qtile = (const char*)(Qh + (size_t)q0 * DQ);
        unsigned ldsbase = (unsigned)(size_t)(&stageQ[0]);   // low 32 bits = LDS offset
        unsigned lo16 = (unsigned)lane * 16u;
        for (int it = 0; it < 4; ++it) {
            unsigned o   = lo16 + (unsigned)it * 512u;
            unsigned lds = ldsbase + o;
            const char* g = qtile + o;
            asm volatile("global_load_async_to_lds_b128 %0, %1, off"
                         :: "v"(lds), "v"(g) : "memory");
        }
        asm volatile("s_wait_asynccnt 0x0" ::: "memory");
    }
    __syncthreads();

    // Q A-fragments from LDS (held for the whole kernel): d-steps 0 and 32
    const bf16_t* qrow = &stageQ[lo * DQ];
    ABFrag a0, a1;
    a0.half8[0] = *(const v8bf*)(qrow +  0 + hi * 16);
    a0.half8[1] = *(const v8bf*)(qrow +  8 + hi * 16);
    a1.half8[0] = *(const v8bf*)(qrow + 32 + hi * 16);
    a1.half8[1] = *(const v8bf*)(qrow + 40 + hi * 16);

    const float scale = 0.125f;             // 1/sqrt(64)
    const int kbase = wid * 256;

    // ---- pass 1: online (max, sumexp) per row over this wave's 256 keys ----
    float mr[8], sr[8];
    for (int r = 0; r < 8; ++r) { mr[r] = -3.0e38f; sr[r] = 0.f; }

    for (int c = 0; c < 16; ++c) {
        const int kk = kbase + c * 16;
        int kpf = kk + 16 > SEQ - 16 ? SEQ - 16 : kk + 16;
        __builtin_prefetch(Kh + (size_t)(kpf + lo) * DQ, 0, 1);
        v16bf b0 = *(const v16bf*)(Kh + (size_t)(kk + lo) * DQ +  0 + hi * 16);
        v16bf b1 = *(const v16bf*)(Kh + (size_t)(kk + lo) * DQ + 32 + hi * 16);
        v8f acc = {0.f,0.f,0.f,0.f,0.f,0.f,0.f,0.f};
        acc = wmma_bf16(a0.v, b0, acc);
        acc = wmma_bf16(a1.v, b1, acc);
        for (int r = 0; r < 8; ++r) {
            float v  = acc[r] * scale;
            float mn = fmaxf(mr[r], v);
            sr[r] = sr[r] * __expf(mr[r] - mn) + __expf(v - mn);
            mr[r] = mn;
        }
    }
    // reduce over the 16 column-lanes of each half-wave
    for (int off = 1; off < 16; off <<= 1) {
        for (int r = 0; r < 8; ++r) {
            float om = __shfl_xor(mr[r], off, 32);
            float os = __shfl_xor(sr[r], off, 32);
            float mn = fmaxf(mr[r], om);
            sr[r] = sr[r] * __expf(mr[r] - mn) + os * __expf(om - mn);
            mr[r] = mn;
        }
    }
    if (lo == 0) {
        for (int r = 0; r < 8; ++r) {
            wave_m[wid][r + hi * 8] = mr[r];
            wave_s[wid][r + hi * 8] = sr[r];
        }
    }
    __syncthreads();
    if (tid < 16) {
        float m = -3.0e38f;
        for (int w = 0; w < 8; ++w) m = fmaxf(m, wave_m[w][tid]);
        float s = 0.f;
        for (int w = 0; w < 8; ++w) s += wave_s[w][tid] * __expf(wave_m[w][tid] - m);
        gM[tid]    = m;
        gRinv[tid] = 1.f / s;
    }
    __syncthreads();

    float myM[8], myR[8];
    for (int r = 0; r < 8; ++r) { myM[r] = gM[r + hi * 8]; myR[r] = gRinv[r + hi * 8]; }

    // ---- pass 2: recompute scores, write attn (NT), accumulate context -----
    v8f ctx[4];
    for (int t = 0; t < 4; ++t) ctx[t] = (v8f){0.f,0.f,0.f,0.f,0.f,0.f,0.f,0.f};

    for (int c = 0; c < 8; ++c) {
        const int kk = kbase + c * 32;
        for (int halfk = 0; halfk < 2; ++halfk) {
            const int kh = kk + halfk * 16;
            v16bf b0 = *(const v16bf*)(Kh + (size_t)(kh + lo) * DQ +  0 + hi * 16);
            v16bf b1 = *(const v16bf*)(Kh + (size_t)(kh + lo) * DQ + 32 + hi * 16);
            v8f acc = {0.f,0.f,0.f,0.f,0.f,0.f,0.f,0.f};
            acc = wmma_bf16(a0.v, b0, acc);
            acc = wmma_bf16(a1.v, b1, acc);
            for (int r = 0; r < 8; ++r) {
                int row = r + hi * 8;
                float p = __expf(acc[r] * scale - myM[r]) * myR[r];
                __builtin_nontemporal_store(p, attnB + (size_t)row * SEQ + kh + lo);
                stage[wid][row][halfk * 16 + lo] = (bf16_t)p;
            }
        }
        // re-layout probabilities C-tile -> A-fragment (16x32) via LDS stage
        ABFrag pa;
        pa.half8[0] = *(const v8bf*)(&stage[wid][lo][ 0 + hi * 16]);
        pa.half8[1] = *(const v8bf*)(&stage[wid][lo][ 8 + hi * 16]);
        // context: ctx[16, d] += P[16, kk..kk+31] @ V[kk..kk+31, d]
        for (int dt = 0; dt < 4; ++dt) {
            v16bf bv = *(const v16bf*)(Vh + (size_t)(dt * 16 + lo) * SEQ + kk + hi * 16);
            ctx[dt] = wmma_bf16(pa.v, bv, ctx[dt]);
        }
    }

    // cross-wave context reduction in LDS (ds_add_f32)
    for (int dt = 0; dt < 4; ++dt)
        for (int r = 0; r < 8; ++r)
            atomicAdd(&ctxLds[r + hi * 8][dt * 16 + lo], ctx[dt][r]);
    __syncthreads();

    // ---- async-DMA the reduced 4KB context tile LDS -> global (wave 0) -----
    // GLOBAL_STORE_ASYNC_FROM_LDS_B128: per-lane 16B LDS -> global
    float* ctxg = ctx_out + ((size_t)b * SEQ + q0) * D_MODEL + h * DQ;
    if (wid == 0) {
        unsigned ldsbase = (unsigned)(size_t)(&ctxLds[0][0]);
        for (int it = 0; it < 8; ++it) {
            int cchunk = it * 32 + lane;            // 256 x 16B chunks
            int rr = cchunk >> 4;
            int d4 = (cchunk & 15) << 2;
            unsigned lds = ldsbase + (unsigned)cchunk * 16u;
            const float* g = ctxg + (size_t)rr * D_MODEL + d4;
            asm volatile("global_store_async_from_lds_b128 %0, %1, off"
                         :: "v"(g), "v"(lds) : "memory");
        }
        asm volatile("s_wait_asynccnt 0x0" ::: "memory");
    }
}

// ---------------- stage 3: out = LayerNorm(context + input_Q) --------------
__global__ void __launch_bounds__(256) ln_add(
    const float* __restrict__ ctx, const float* __restrict__ inQ, float* __restrict__ out)
{
    __shared__ float red[2][8];
    const int row = blockIdx.x;
    const float* c = ctx + (size_t)row * D_MODEL;
    const float* x = inQ + (size_t)row * D_MODEL;
    float* o = out + (size_t)row * D_MODEL;
    const int tid = threadIdx.x;

    float v[4], sum = 0.f, sumsq = 0.f;
    for (int i = 0; i < 4; ++i) {
        float t = c[tid + i * 256] + x[tid + i * 256];
        v[i] = t; sum += t; sumsq += t * t;
    }
    for (int off = 16; off; off >>= 1) {
        sum   += __shfl_xor(sum,   off, 32);
        sumsq += __shfl_xor(sumsq, off, 32);
    }
    const int wid = tid >> 5, lane = tid & 31;
    if (lane == 0) { red[0][wid] = sum; red[1][wid] = sumsq; }
    __syncthreads();
    if (tid == 0) {
        float s = 0.f, q = 0.f;
        for (int w = 0; w < 8; ++w) { s += red[0][w]; q += red[1][w]; }
        red[0][0] = s; red[1][0] = q;
    }
    __syncthreads();
    float mu   = red[0][0] * (1.f / D_MODEL);
    float var  = red[1][0] * (1.f / D_MODEL) - mu * mu;
    float rinv = rsqrtf(var + LN_EPS);
    for (int i = 0; i < 4; ++i) o[tid + i * 256] = (v[i] - mu) * rinv;
}

// ---------------- launcher ----------------
extern "C" void kernel_launch(void* const* d_in, const int* in_sizes, int n_in,
                              void* d_out, int out_size, void* d_ws, size_t ws_size,
                              hipStream_t stream) {
    const float* inQ = (const float*)d_in[0];
    const float* inK = (const float*)d_in[1];
    const float* inV = (const float*)d_in[2];
    const float* wQ  = (const float*)d_in[3];
    const float* wK  = (const float*)d_in[4];
    const float* wV  = (const float*)d_in[5];

    const size_t NX = (size_t)MROWS * D_MODEL;      // 4,194,304 elems
    const size_t NW = (size_t)D_MODEL * D_MODEL;    // 1,048,576 elems

    char* w = (char*)d_ws;
    auto carve = [&](size_t bytes) {
        void* p = (void*)w;
        w += (bytes + 255) & ~(size_t)255;
        return p;
    };
    bf16_t* Xq = (bf16_t*)carve(NX * 2);
    bf16_t* Xk = (bf16_t*)carve(NX * 2);
    bf16_t* Xv = (bf16_t*)carve(NX * 2);
    bf16_t* Wq = (bf16_t*)carve(NW * 2);
    bf16_t* Wk = (bf16_t*)carve(NW * 2);
    bf16_t* Wv = (bf16_t*)carve(NW * 2);
    bf16_t* Qp = (bf16_t*)carve(NX * 2);
    bf16_t* Kp = (bf16_t*)carve(NX * 2);
    bf16_t* Vt = (bf16_t*)carve(NX * 2);
    float*  Ctx = (float*)carve(NX * 4);

    float* outLN = (float*)d_out;
    float* attnO = (float*)d_out + NX;   // attn follows `out` in return order

    cvt_f32_bf16<<<2048, 256, 0, stream>>>(inQ, Xq, (int)NX);
    cvt_f32_bf16<<<2048, 256, 0, stream>>>(inK, Xk, (int)NX);
    cvt_f32_bf16<<<2048, 256, 0, stream>>>(inV, Xv, (int)NX);
    cvt_f32_bf16<<<1024, 256, 0, stream>>>(wQ, Wq, (int)NW);
    cvt_f32_bf16<<<1024, 256, 0, stream>>>(wK, Wk, (int)NW);
    cvt_f32_bf16<<<1024, 256, 0, stream>>>(wV, Wv, (int)NW);

    proj_wmma<<<dim3(32, 64, 3), 256, 0, stream>>>(Xq, Xk, Xv, Wq, Wk, Wv, Qp, Kp, Vt);

    attn_wmma<<<dim3(BATCH * N_HEADS * (SEQ / 16)), 256, 0, stream>>>(Qp, Kp, Vt, attnO, Ctx);

    ln_add<<<dim3(MROWS), 256, 0, stream>>>(Ctx, inQ, outLN);
}
